// RBFMultiHeadAttn_55722905699262
// MI455X (gfx1250) — matrix operand order repaired and checked by reading, not compiled, not run on hardware
//
#include <hip/hip_runtime.h>

#define S_LEN 1024
#define N_B   8
#define N_H   8
#define D_H   64
#define D_M   512
#define BH_   (N_B * N_H)      // 64
#define SB_   (S_LEN * N_B)    // 8192
#define SCALE_F 0.125f
#define LOG_A   (-1.6094379124341003f)  // log(0.2)
#define LN_EPS  1e-5f

typedef __attribute__((ext_vector_type(16))) __bf16 v16bf;
typedef __attribute__((ext_vector_type(8)))  __bf16 v8bf;
typedef __attribute__((ext_vector_type(8)))  float  v8f;

static __device__ __forceinline__ int lane_id() { return threadIdx.x & 31; }

// A fragment (16 rows x 32 k, bf16), row-major source base[row*ld + k].
// Layout: M=lane%16, element s = K 16hi+s (contiguous, matches load_fragB)
static __device__ __forceinline__ v16bf load_fragA(const __bf16* base, int ld) {
  int l = lane_id();
  int m = l & 15, hi = l >> 4;
  const __bf16* r = base + m * ld;
  return *(const v16bf*)(r + 16 * hi);
}

// B fragment (32 k x 16 n, bf16) where B[k][n] = base[n*ld + k]
// Layout: lane<16: N=lane, K=0..15 ; lane>=16: N=lane-16, K=16..31 (contiguous 32B per lane)
static __device__ __forceinline__ v16bf load_fragB(const __bf16* base, int ld) {
  int l = lane_id();
  int n = l & 15, hi = l >> 4;
  return *(const v16bf*)(base + n * ld + 16 * hi);
}

static __device__ __forceinline__ v8f wmma_bf16(v16bf a, v16bf b, v8f c) {
  return __builtin_amdgcn_wmma_f32_16x16x32_bf16(false, a, false, b, (short)0, c, false, false);
}

// reduce over the 16 lanes of a half-wave (columns of a D tile)
static __device__ __forceinline__ float half_reduce16(float v) {
#pragma unroll
  for (int m = 1; m < 16; m <<= 1) v += __shfl_xor(v, m, 32);
  return v;
}

// ---------------- Kernel 0: fp32 -> bf16 conversion of h and weights ----------------
__global__ __launch_bounds__(256)
void k_convert(const float* __restrict__ h, const float* __restrict__ Wq,
               const float* __restrict__ Wkv, const float* __restrict__ Wo,
               __bf16* __restrict__ hb, __bf16* __restrict__ Wc, __bf16* __restrict__ Wob) {
  int i = blockIdx.x * blockDim.x + threadIdx.x;
  const int nh  = SB_ * D_M;       // 4194304
  const int nwq = D_M * D_M;       // 262144
  const int nwkv = 2 * D_M * D_M;  // 524288
  if (i < nh)   hb[i] = (__bf16)h[i];
  if (i < nwq)  Wc[i] = (__bf16)Wq[i];
  if (i < nwkv) Wc[nwq + i] = (__bf16)Wkv[i];
  if (i < nwq)  Wob[i] = (__bf16)Wo[i];
}

// ---------------- Kernel 1: fused QKV projection GEMM + per-head pack ----------------
// O[r][n] = sum_k hb[r][k] * Wc[n][k], M=8192, N=1536, K=512.
// Columns [0,512)=Q, [512,1024)=K (L2-normalized rows), [1024,1536)=V.
// Software-pipelined: fragments for slice kk+32 are in flight across slice kk's WMMAs.
__global__ __launch_bounds__(256)
void k_qkv(const __bf16* __restrict__ hb, const __bf16* __restrict__ Wc,
           __bf16* __restrict__ Qbh, __bf16* __restrict__ Kbh,
           __bf16* __restrict__ Vbh, __bf16* __restrict__ Vt,
           float* __restrict__ q2, float* __restrict__ k2, float* __restrict__ v2) {
  int w = threadIdx.x >> 5;
  int l = lane_id();
  int n = l & 15, hi = l >> 4;
  int rbase = blockIdx.x * 128 + w * 16;
  int n0 = blockIdx.y * 64;
  const __bf16* Abase = hb + rbase * D_M;
  v8f acc[4] = {};
  v16bf a = load_fragA(Abase, D_M);
  v16bf b[4];
#pragma unroll
  for (int t = 0; t < 4; ++t) b[t] = load_fragB(Wc + (n0 + 16 * t) * D_M, D_M);
#pragma unroll 4
  for (int kk = 0; kk < D_M; kk += 32) {
    int kn = (kk + 32 < D_M) ? (kk + 32) : kk;   // clamped prefetch index (branch-free)
    v16bf an = load_fragA(Abase + kn, D_M);
    v16bf bn[4];
#pragma unroll
    for (int t = 0; t < 4; ++t) bn[t] = load_fragB(Wc + (n0 + 16 * t) * D_M + kn, D_M);
#pragma unroll
    for (int t = 0; t < 4; ++t) acc[t] = wmma_bf16(a, b[t], acc[t]);
    a = an;
#pragma unroll
    for (int t = 0; t < 4; ++t) b[t] = bn[t];
  }
  int sect = n0 >> 9;           // 0=Q, 1=K, 2=V  (block-uniform)
  int head = (n0 >> 6) & 7;
#pragma unroll
  for (int e = 0; e < 8; ++e) {
    int m = e + 8 * hi;
    int r = rbase + m;
    int s = r >> 3, bb = r & 7;
    int bh = bb * N_H + head;
    if (sect == 1) {
      // L2-normalize this K row (64 wide across 4 accs x 16 lanes of half-wave)
      float p = 0.f;
#pragma unroll
      for (int t = 0; t < 4; ++t) p += acc[t][e] * acc[t][e];
      p = half_reduce16(p);
      float sc = 1.0f / fmaxf(sqrtf(p), 1e-12f);
      float p2 = 0.f;
#pragma unroll
      for (int t = 0; t < 4; ++t) {
        __bf16 kb = (__bf16)(acc[t][e] * sc);
        Kbh[((size_t)bh * S_LEN + s) * D_H + t * 16 + n] = kb;
        float kf = (float)kb;
        p2 += kf * kf;
      }
      p2 = half_reduce16(p2);
      if (n == 0) k2[bh * S_LEN + s] = p2;
    } else {
      float p2 = 0.f;
#pragma unroll
      for (int t = 0; t < 4; ++t) {
        __bf16 vb = (__bf16)acc[t][e];
        float vf = (float)vb;
        p2 += vf * vf;
        int d = t * 16 + n;
        if (sect == 0) {
          Qbh[((size_t)bh * S_LEN + s) * D_H + d] = vb;
        } else {
          Vbh[((size_t)bh * S_LEN + s) * D_H + d] = vb;
          Vt[((size_t)bh * D_H + d) * S_LEN + s] = vb;   // transposed copy for P@V
        }
      }
      p2 = half_reduce16(p2);
      if (n == 0) {
        if (sect == 0) q2[bh * S_LEN + s] = p2;
        else           v2[bh * S_LEN + s] = p2;
      }
    }
  }
}

// ---------------- Kernel 2: RBF row sums for k and kv weights ----------------
// rsk[i]  = sum_j exp(-SCALE * max(k2i+k2j-2 K_i.K_j, 0))
// rskv[i] = sum_j exp(-SCALE * max((k2i+k2j-2Kij) + (v2i+v2j-2Vij), 0))
// Pipelined: next tile's fragments load across current tile's WMMAs + exp epilogue.
__global__ __launch_bounds__(256)
void k_rowsums(const __bf16* __restrict__ Kbh, const __bf16* __restrict__ Vbh,
               const float* __restrict__ k2, const float* __restrict__ v2,
               float* __restrict__ rsk, float* __restrict__ rskv) {
  __shared__ float lsk[8][16];
  __shared__ float lskv[8][16];
  int w = threadIdx.x >> 5;
  int l = lane_id();
  int n = l & 15, hi = l >> 4;
  int bh = blockIdx.y;
  int i0 = blockIdx.x * 16;
  const __bf16* Kb = Kbh + (size_t)bh * S_LEN * D_H;
  const __bf16* Vb = Vbh + (size_t)bh * S_LEN * D_H;
  const float* k2b = k2 + bh * S_LEN;
  const float* v2b = v2 + bh * S_LEN;
  v16bf aK0 = load_fragA(Kb + i0 * D_H, D_H);
  v16bf aK1 = load_fragA(Kb + i0 * D_H + 32, D_H);
  v16bf aV0 = load_fragA(Vb + i0 * D_H, D_H);
  v16bf aV1 = load_fragA(Vb + i0 * D_H + 32, D_H);
  float k2i[8], v2i[8];
#pragma unroll
  for (int e = 0; e < 8; ++e) {
    int i = i0 + e + 8 * hi;
    k2i[e] = k2b[i]; v2i[e] = v2b[i];
  }
  float accK[8] = {};
  float accKV[8] = {};
  // prologue loads for first tile
  int j0 = w * 16;
  v16bf bK0 = load_fragB(Kb + j0 * D_H, D_H);
  v16bf bK1 = load_fragB(Kb + j0 * D_H + 32, D_H);
  v16bf bV0 = load_fragB(Vb + j0 * D_H, D_H);
  v16bf bV1 = load_fragB(Vb + j0 * D_H + 32, D_H);
  float k2j = k2b[j0 + n];
  float v2j = v2b[j0 + n];
  for (int jt = w; jt < 64; jt += 8) {
    int jn = ((jt + 8) < 64) ? (jt + 8) * 16 : jt * 16;   // clamped prefetch
    v16bf nK0 = load_fragB(Kb + jn * D_H, D_H);
    v16bf nK1 = load_fragB(Kb + jn * D_H + 32, D_H);
    v16bf nV0 = load_fragB(Vb + jn * D_H, D_H);
    v16bf nV1 = load_fragB(Vb + jn * D_H + 32, D_H);
    float nk2 = k2b[jn + n];
    float nv2 = v2b[jn + n];
    v8f sk = {}; sk = wmma_bf16(aK0, bK0, sk); sk = wmma_bf16(aK1, bK1, sk);
    v8f sv = {}; sv = wmma_bf16(aV0, bV0, sv); sv = wmma_bf16(aV1, bV1, sv);
#pragma unroll
    for (int e = 0; e < 8; ++e) {
      float d2k = k2i[e] + k2j - 2.0f * sk[e];
      float d2v = v2i[e] + v2j - 2.0f * sv[e];
      accK[e]  += __expf(-SCALE_F * fmaxf(d2k, 0.0f));
      accKV[e] += __expf(-SCALE_F * fmaxf(d2k + d2v, 0.0f));
    }
    bK0 = nK0; bK1 = nK1; bV0 = nV0; bV1 = nV1;
    k2j = nk2; v2j = nv2;
  }
#pragma unroll
  for (int e = 0; e < 8; ++e) {
    accK[e]  = half_reduce16(accK[e]);
    accKV[e] = half_reduce16(accKV[e]);
  }
  if (n == 0) {
#pragma unroll
    for (int e = 0; e < 8; ++e) {
      lsk[w][e + 8 * hi]  = accK[e];
      lskv[w][e + 8 * hi] = accKV[e];
    }
  }
  __syncthreads();
  int t = threadIdx.x;
  if (t < 16) {
    float a = 0.f, b = 0.f;
#pragma unroll
    for (int ww = 0; ww < 8; ++ww) { a += lsk[ww][t]; b += lskv[ww][t]; }
    rsk[bh * S_LEN + i0 + t]  = a;
    rskv[bh * S_LEN + i0 + t] = b;
  }
}

// ---------------- Kernel 3: log weights (huber + logsumexp), one block per (b,h) ----------------
__global__ __launch_bounds__(256)
void k_weights(const float* __restrict__ rsk, const float* __restrict__ rskv,
               float* __restrict__ kw, float* __restrict__ kvw) {
  __shared__ float red[256];
  __shared__ float sm_k, sm_v, sz_k, sz_v;
  int bh = blockIdx.x, t = threadIdx.x;
  const float* rk = rsk + bh * S_LEN;
  const float* rv = rskv + bh * S_LEN;
  const float invS = 1.0f / (float)S_LEN;
  // obj3 = mean_j obj2[j] = (sum rs) / S^2
  float a = 0.f, b = 0.f;
  for (int i = t; i < S_LEN; i += 256) { a += rk[i]; b += rv[i]; }
  red[t] = a; __syncthreads();
  for (int s = 128; s > 0; s >>= 1) { if (t < s) red[t] += red[t + s]; __syncthreads(); }
  if (t == 0) sm_k = red[0] * invS * invS;
  __syncthreads();
  red[t] = b; __syncthreads();
  for (int s = 128; s > 0; s >>= 1) { if (t < s) red[t] += red[t + s]; __syncthreads(); }
  if (t == 0) sm_v = red[0] * invS * invS;
  __syncthreads();
  float o3k = sm_k, o3v = sm_v;
  // Z = sum_i exp(log_norm_i) = sum_i 0.2 / sqrt(1 + obj2_i + obj3)
  float za = 0.f, zb = 0.f;
  for (int i = t; i < S_LEN; i += 256) {
    za += 0.2f * rsqrtf(1.0f + rk[i] * invS + o3k);
    zb += 0.2f * rsqrtf(1.0f + rv[i] * invS + o3v);
  }
  red[t] = za; __syncthreads();
  for (int s = 128; s > 0; s >>= 1) { if (t < s) red[t] += red[t + s]; __syncthreads(); }
  if (t == 0) sz_k = logf(red[0]);
  __syncthreads();
  red[t] = zb; __syncthreads();
  for (int s = 128; s > 0; s >>= 1) { if (t < s) red[t] += red[t + s]; __syncthreads(); }
  if (t == 0) sz_v = logf(red[0]);
  __syncthreads();
  float lzk = sz_k, lzv = sz_v;
  for (int i = t; i < S_LEN; i += 256) {
    kw[bh * S_LEN + i]  = LOG_A - 0.5f * logf(1.0f + rk[i] * invS + o3k) - lzk;
    kvw[bh * S_LEN + i] = LOG_A - 0.5f * logf(1.0f + rv[i] * invS + o3v) - lzv;
  }
}

// ---------------- Kernel 4: RBF attention (scores + P@V), flash-style ----------------
// All 8 global fragment loads for a chunk are issued up front, so the V^T loads are
// in flight across the score WMMAs, exp epilogue, and the LDS barrier.
__global__ __launch_bounds__(256)
void k_attn(const __bf16* __restrict__ Qbh, const __bf16* __restrict__ Kbh,
            const __bf16* __restrict__ Vt,
            const float* __restrict__ q2, const float* __restrict__ k2,
            const float* __restrict__ kw, const float* __restrict__ kvw,
            __bf16* __restrict__ avb) {
  __shared__ __align__(32) __bf16 Pl[8][16 * 32];  // per-wave P tile (16 i x 32 j)
  __shared__ float numl[8][16 * 64];
  __shared__ float denl[8][16];
  __shared__ float denf[16];
  int w = threadIdx.x >> 5;
  int l = lane_id();
  int n = l & 15, hi = l >> 4;
  int bh = blockIdx.y;
  int i0 = blockIdx.x * 16;
  const __bf16* Qb  = Qbh + (size_t)bh * S_LEN * D_H;
  const __bf16* Kb  = Kbh + (size_t)bh * S_LEN * D_H;
  const __bf16* Vtb = Vt + (size_t)bh * D_H * S_LEN;
  const float* k2b = k2 + bh * S_LEN;
  const float* kwb = kw + bh * S_LEN;
  const float* kvb = kvw + bh * S_LEN;
  v16bf aQ0 = load_fragA(Qb + i0 * D_H, D_H);
  v16bf aQ1 = load_fragA(Qb + i0 * D_H + 32, D_H);
  float q2i[8];
#pragma unroll
  for (int e = 0; e < 8; ++e) q2i[e] = q2[bh * S_LEN + i0 + e + 8 * hi];
  v8f vacc[4] = {};
  float den[8] = {};
  for (int jc = w; jc < 32; jc += 8) {   // 32-wide j chunks; 4 per wave (uniform)
    int j0 = jc * 32;
    // issue every global fragment load for this chunk first
    v16bf bK[4];
    bK[0] = load_fragB(Kb + j0 * D_H, D_H);
    bK[1] = load_fragB(Kb + j0 * D_H + 32, D_H);
    bK[2] = load_fragB(Kb + (j0 + 16) * D_H, D_H);
    bK[3] = load_fragB(Kb + (j0 + 16) * D_H + 32, D_H);
    v16bf bV[4];
#pragma unroll
    for (int t = 0; t < 4; ++t) bV[t] = load_fragB(Vtb + (t * 16) * S_LEN + j0, S_LEN);
#pragma unroll
    for (int jt = 0; jt < 2; ++jt) {
      int jb = j0 + jt * 16;
      v8f sc = {};
      sc = wmma_bf16(aQ0, bK[jt * 2 + 0], sc);
      sc = wmma_bf16(aQ1, bK[jt * 2 + 1], sc);
      float k2j = k2b[jb + n];
      float kwj = kwb[jb + n];
      float kvj = kvb[jb + n];
#pragma unroll
      for (int e = 0; e < 8; ++e) {
        float s_ = -SCALE_F * fmaxf(q2i[e] + k2j - 2.0f * sc[e], 0.0f);
        den[e] += __expf(kwj + s_);
        Pl[w][(e + 8 * hi) * 32 + jt * 16 + n] = (__bf16)__expf(kvj + s_);
      }
    }
    __syncthreads();   // uniform trip count: P visible for fragment gathers
    v16bf aP = load_fragA(&Pl[w][0], 32);
#pragma unroll
    for (int t = 0; t < 4; ++t) vacc[t] = wmma_bf16(aP, bV[t], vacc[t]);
    __syncthreads();   // before next chunk overwrites Pl
  }
  // cross-wave reduction of numerator / denominator
#pragma unroll
  for (int e = 0; e < 8; ++e) den[e] = half_reduce16(den[e]);
  if (n == 0) {
#pragma unroll
    for (int e = 0; e < 8; ++e) denl[w][e + 8 * hi] = den[e];
  }
#pragma unroll
  for (int t = 0; t < 4; ++t)
#pragma unroll
    for (int e = 0; e < 8; ++e)
      numl[w][(e + 8 * hi) * 64 + t * 16 + n] = vacc[t][e];
  __syncthreads();
  int tid = threadIdx.x;
  if (tid < 16) {
    float d_ = 0.f;
#pragma unroll
    for (int ww = 0; ww < 8; ++ww) d_ += denl[ww][tid];
    denf[tid] = d_;
  }
  __syncthreads();
  int bb = bh >> 3, head = bh & 7;
#pragma unroll
  for (int q = 0; q < 4; ++q) {
    int flat = tid * 4 + q;
    int m = flat >> 6, d = flat & 63;
    float s_ = 0.f;
#pragma unroll
    for (int ww = 0; ww < 8; ++ww) s_ += numl[ww][flat];
    float o = s_ / denf[m];
    avb[((i0 + m) * N_B + bb) * D_M + head * 64 + d] = (__bf16)o;
  }
}

// ---------------- Kernel 5: output GEMM + residual + LayerNorm ----------------
__global__ __launch_bounds__(256)
void k_out(const __bf16* __restrict__ avb, const __bf16* __restrict__ Wob,
           const float* __restrict__ h, const float* __restrict__ lng,
           const float* __restrict__ lnb, float* __restrict__ out) {
  __shared__ float X[16][D_M];   // 32 KB
  __shared__ float red[16][16];
  int w = threadIdx.x >> 5;
  int l = lane_id();
  int n = l & 15, hi = l >> 4;
  int rbase = blockIdx.x * 16;
  int n0 = w * 64;
  const __bf16* Abase = avb + rbase * D_M;
  v8f acc[4] = {};
  v16bf a = load_fragA(Abase, D_M);
  v16bf b[4];
#pragma unroll
  for (int t = 0; t < 4; ++t) b[t] = load_fragB(Wob + (n0 + 16 * t) * D_M, D_M);
#pragma unroll 4
  for (int kk = 0; kk < D_M; kk += 32) {
    int kn = (kk + 32 < D_M) ? (kk + 32) : kk;
    v16bf an = load_fragA(Abase + kn, D_M);
    v16bf bn[4];
#pragma unroll
    for (int t = 0; t < 4; ++t) bn[t] = load_fragB(Wob + (n0 + 16 * t) * D_M + kn, D_M);
#pragma unroll
    for (int t = 0; t < 4; ++t) acc[t] = wmma_bf16(a, b[t], acc[t]);
    a = an;
#pragma unroll
    for (int t = 0; t < 4; ++t) b[t] = bn[t];
  }
#pragma unroll
  for (int e = 0; e < 8; ++e) {
    int m = e + 8 * hi;
#pragma unroll
    for (int t = 0; t < 4; ++t) {
      int col = n0 + t * 16 + n;
      X[m][col] = acc[t][e] + h[(rbase + m) * D_M + col];
    }
  }
  __syncthreads();
  int row = threadIdx.x >> 4, c = threadIdx.x & 15;
  float sm = 0.f;
  for (int q = 0; q < 32; ++q) sm += X[row][c * 32 + q];
  red[row][c] = sm; __syncthreads();
  float mu = 0.f;
  for (int q = 0; q < 16; ++q) mu += red[row][q];
  mu *= (1.0f / (float)D_M);
  __syncthreads();
  float vs = 0.f;
  for (int q = 0; q < 32; ++q) { float d = X[row][c * 32 + q] - mu; vs += d * d; }
  red[row][c] = vs; __syncthreads();
  float var = 0.f;
  for (int q = 0; q < 16; ++q) var += red[row][q];
  var *= (1.0f / (float)D_M);
  float inv = rsqrtf(var + LN_EPS);
  for (int q = 0; q < 32; ++q) {
    int col = c * 32 + q;
    out[(rbase + row) * D_M + col] = (X[row][col] - mu) * inv * lng[col] + lnb[col];
  }
}

extern "C" void kernel_launch(void* const* d_in, const int* in_sizes, int n_in,
                              void* d_out, int out_size, void* d_ws, size_t ws_size,
                              hipStream_t stream) {
  (void)in_sizes; (void)n_in; (void)out_size; (void)ws_size;
  const float* h   = (const float*)d_in[0];
  const float* Wq  = (const float*)d_in[1];
  const float* Wkv = (const float*)d_in[2];
  const float* Wo  = (const float*)d_in[3];
  const float* lng = (const float*)d_in[4];
  const float* lnb = (const float*)d_in[5];
  float* out = (float*)d_out;

  char* ws = (char*)d_ws;
  size_t off = 0;
  auto take = [&](size_t bytes) -> void* {
    void* p = ws + off;
    off += (bytes + 255) & ~(size_t)255;
    return p;
  };
  __bf16* hb  = (__bf16*)take((size_t)SB_ * D_M * 2);        // 8 MB
  __bf16* Wc  = (__bf16*)take((size_t)1536 * 512 * 2);       // 1.5 MB
  __bf16* Wob = (__bf16*)take((size_t)512 * 512 * 2);        // 0.5 MB
  __bf16* Qbh = (__bf16*)take((size_t)BH_ * S_LEN * D_H * 2);
  __bf16* Kbh = (__bf16*)take((size_t)BH_ * S_LEN * D_H * 2);
  __bf16* Vbh = (__bf16*)take((size_t)BH_ * S_LEN * D_H * 2);
  __bf16* Vt  = (__bf16*)take((size_t)BH_ * S_LEN * D_H * 2);
  __bf16* avb = (__bf16*)take((size_t)SB_ * D_M * 2);        // 8 MB
  float* q2   = (float*)take((size_t)BH_ * S_LEN * 4);
  float* k2   = (float*)take((size_t)BH_ * S_LEN * 4);
  float* v2   = (float*)take((size_t)BH_ * S_LEN * 4);
  float* rsk  = (float*)take((size_t)BH_ * S_LEN * 4);
  float* rskv = (float*)take((size_t)BH_ * S_LEN * 4);
  float* kwp  = (float*)take((size_t)BH_ * S_LEN * 4);
  float* kvwp = (float*)take((size_t)BH_ * S_LEN * 4);

  k_convert<<<(SB_ * D_M) / 256, 256, 0, stream>>>(h, Wq, Wkv, Wo, hb, Wc, Wob);
  k_qkv<<<dim3(SB_ / 128, 1536 / 64), 256, 0, stream>>>(hb, Wc, Qbh, Kbh, Vbh, Vt, q2, k2, v2);
  k_rowsums<<<dim3(S_LEN / 16, BH_), 256, 0, stream>>>(Kbh, Vbh, k2, v2, rsk, rskv);
  k_weights<<<BH_, 256, 0, stream>>>(rsk, rskv, kwp, kvwp);
  k_attn<<<dim3(S_LEN / 16, BH_), 256, 0, stream>>>(Qbh, Kbh, Vt, q2, k2, kwp, kvwp, avb);
  k_out<<<SB_ / 16, 256, 0, stream>>>(avb, Wob, h, lng, lnb, out);
}
